// DeformableConv2d_21998822490385
// MI455X (gfx1250) — matrix operand order repaired and checked by reading, not compiled, not run on hardware
//
#include <hip/hip_runtime.h>
#include <hip/hip_bf16.h>

// DCNv2 deformable conv for MI455X (gfx1250, wave32, WMMA).
// B=4, C=64, O=64, H=W=128, KK=9. Main contraction is a GEMM
// M=65536, N=64, K=576 done with v_wmma_f32_16x16x32_f16, fused with the
// bilinear gather (im2col staged in LDS, never written to HBM).
// Block = 256 threads = 8 wave32, owns 64 pixels x 64 outputs:
// 16 WMMA tiles, 2 per wave -> B fragments reused across 2 WMMAs.

typedef __attribute__((ext_vector_type(16))) _Float16 v16h;
typedef __attribute__((ext_vector_type(8)))  float    v8f;

#define HH 128
#define WW 128
#define HWSZ 16384          // H*W
#define CCH 64
#define OCH 64
#define NCHUNK 18           // K = 576 = 18 chunks of 32, kdim = k*64 + c
#define AROW 40             // LDS A row stride in halves (80B, 16B multiple)
#define PXB 64              // pixels per block

// ---------------------------------------------------------------------------
// Kernel 1: param-generating 3x3 conv (27 channels) + bias + sigmoid(mask).
// One thread per pixel; pg_weight cached in LDS as [tap][c][28] (padded so
// the 27-wide j-run is 16B aligned -> ds_load_b128). Writes per-pixel [27]
// floats: ch 2k = dy_k, ch 2k+1 = dx_k (the reference's interleaved reshape
// of concat(oh,ow)), ch 18+k = sigmoid(mask_k).
// ---------------------------------------------------------------------------
__global__ __launch_bounds__(256) void pg_conv_kernel(
    const float* __restrict__ x, const float* __restrict__ pgw,
    const float* __restrict__ pgb, float* __restrict__ offm)
{
    __shared__ __align__(16) float sw[9 * 64 * 28];
    __shared__ float sb[27];
    const int tid = threadIdx.x;
    for (int i = tid; i < 9 * 64 * 27; i += 256) {
        int j = i % 27, c = (i / 27) % 64, t = i / (27 * 64);
        sw[(t * 64 + c) * 28 + j] = pgw[(j * 64 + c) * 9 + t];  // pgw[j][c][t]
    }
    if (tid < 27) sb[tid] = pgb[tid];
    __syncthreads();

    const int p  = blockIdx.x * 256 + tid;
    const int b  = p >> 14;
    const int hw = p & (HWSZ - 1);
    const int h  = hw >> 7, w = hw & 127;

    float acc[27];
#pragma unroll
    for (int j = 0; j < 27; ++j) acc[j] = sb[j];

    const float* xb = x + ((size_t)b * CCH << 14);
    for (int t = 0; t < 9; ++t) {
        const int ty = h - 1 + t / 3, tx = w - 1 + t % 3;
        if (ty < 0 || ty >= HH || tx < 0 || tx >= WW) continue;
        const float* xp  = xb + ty * WW + tx;
        const float* swt = &sw[t * 64 * 28];
        for (int c = 0; c < 64; ++c) {
            const float xv = xp[(size_t)c << 14];
            const float* wc = swt + c * 28;
#pragma unroll
            for (int j = 0; j < 27; ++j) acc[j] = fmaf(xv, wc[j], acc[j]);
        }
    }
    float* op = offm + (size_t)p * 27;
#pragma unroll
    for (int j = 0; j < 27; ++j) {
        float v = acc[j];
        if (j >= 18) v = 1.0f / (1.0f + __expf(-v));   // sigmoid mask
        op[j] = v;
    }
}

// ---------------------------------------------------------------------------
// Kernel 2: pack weight (O,C,3,3) into f16 B-fragments, fragment-major:
// bfrag[((q*4 + ntile)*32 + lane)*16 + half]. CDNA5 B (32x16 f16) lane
// layout used here (contiguous k, same as the A reads; not the ISA table order): lane<16 -> N=lane, K=half(0..15);
// lane>=16 -> N=lane-16, K=16+half. kdim = k*64 + c.
// ---------------------------------------------------------------------------
__global__ __launch_bounds__(256) void pack_b_kernel(
    const float* __restrict__ wgt, _Float16* __restrict__ bfrag)
{
    const int tid = blockIdx.x * 256 + threadIdx.x;
    if (tid >= NCHUNK * 4 * 32 * 16) return;
    const int j  = tid & 15;
    const int l  = (tid >> 4) & 31;
    const int nt = (tid >> 9) & 3;
    const int q  = tid >> 11;
    const int o  = nt * 16 + (l & 15);
    const int kl = ((l >> 4) << 4) | j;     // 0..31 within chunk
    const int kd = q * 32 + kl;             // kdim = k*64 + c
    const int k  = kd >> 6, c = kd & 63;
    bfrag[tid] = (_Float16)wgt[(o * 64 + c) * 9 + k];
}

// ---------------------------------------------------------------------------
// Kernel 3: fused bilinear gather + WMMA GEMM.
// ---------------------------------------------------------------------------
__global__ __launch_bounds__(256) void dcn_wmma_kernel(
    const float* __restrict__ x, const float* __restrict__ offm,
    const _Float16* __restrict__ bfrag, float* __restrict__ out)
{
    __shared__ __align__(16) float s_off[PXB * 27];       // 6912 B, async-staged
    __shared__ int   s_idx[PXB * 9 * 4];                  // 64 px * 9 taps * 4
    __shared__ float s_wt [PXB * 9 * 4];
    __shared__ __align__(16) _Float16 s_a[2][PXB * AROW]; // double-buffered A

    const int tid   = threadIdx.x;
    const int mbase = blockIdx.x * PXB;                   // 16384 % 64 == 0
    const int b     = mbase >> 14;
    const float* xb = x + ((size_t)b * CCH << 14);

    // Phase 0: bulk-copy this block's offsets/masks into LDS.
    // gfx1250: GLOBAL_LOAD_ASYNC_TO_LDS_B128 (ASYNCcnt) - one 16B chunk per
    // active lane per issue; 432 chunks cover 64px * 27 floats.
    {
        const float* gsrc = offm + (size_t)mbase * 27;
#if defined(__gfx1250__)
        for (int i = tid; i < (PXB * 27) / 4; i += 256) {
            unsigned           lds = (unsigned)(size_t)(&s_off[i * 4]);
            unsigned long long ga  = (unsigned long long)(size_t)(gsrc + i * 4);
            asm volatile("global_load_async_to_lds_b128 %0, %1, off"
                         :: "v"(lds), "v"(ga) : "memory");
        }
        asm volatile("s_wait_asynccnt 0x0" ::: "memory");
#else
        for (int i = tid; i < PXB * 27; i += 256) s_off[i] = gsrc[i];
#endif
    }
    __syncthreads();

    // Phase 1: bilinear metadata, once per (pixel, tap); reused over 64 ch.
    for (int e = tid; e < PXB * 9; e += 256) {
        const int pp = e / 9, k = e % 9;
        const int p  = mbase + pp;
        const int hw = p & (HWSZ - 1);
        const int h  = hw >> 7, w = hw & 127;
        const float* om = &s_off[pp * 27];
        const float dy = om[2 * k], dx = om[2 * k + 1], m = om[18 + k];
        const float py = (float)(h - 1 + k / 3) + dy;
        const float px = (float)(w - 1 + k % 3) + dx;
        const float y0f = floorf(py), x0f = floorf(px);
        const float ly = py - y0f, lx = px - x0f;
        const int y0 = (int)y0f, x0 = (int)x0f, y1 = y0 + 1, x1 = x0 + 1;
        const float vy0 = (y0 >= 0 && y0 < HH) ? 1.f : 0.f;
        const float vy1 = (y1 >= 0 && y1 < HH) ? 1.f : 0.f;
        const float vx0 = (x0 >= 0 && x0 < WW) ? 1.f : 0.f;
        const float vx1 = (x1 >= 0 && x1 < WW) ? 1.f : 0.f;
        const int cy0 = min(max(y0, 0), HH - 1), cy1 = min(max(y1, 0), HH - 1);
        const int cx0 = min(max(x0, 0), WW - 1), cx1 = min(max(x1, 0), WW - 1);
        s_idx[e * 4 + 0] = cy0 * WW + cx0;
        s_idx[e * 4 + 1] = cy0 * WW + cx1;
        s_idx[e * 4 + 2] = cy1 * WW + cx0;
        s_idx[e * 4 + 3] = cy1 * WW + cx1;
        s_wt [e * 4 + 0] = (1.f - ly) * (1.f - lx) * vy0 * vx0 * m;
        s_wt [e * 4 + 1] = (1.f - ly) * lx * vy0 * vx1 * m;
        s_wt [e * 4 + 2] = ly * (1.f - lx) * vy1 * vx0 * m;
        s_wt [e * 4 + 3] = ly * lx * vy1 * vx1 * m;
    }
    __syncthreads();

    // Producer: stage 64px x 32ch f16 A tile for chunk q (tap k = q>>1 fixed,
    // so the 4 gather addresses/weights are chunk-invariant per pixel).
    auto produce = [&](int q, int buf) {
        const int k  = q >> 1;
        const int c0 = (q & 1) << 5;
        const int pp = tid >> 2;                // 64 pixels
        const int cl = (tid & 3) << 3;          // 8 channels per thread
        const int e4 = (pp * 9 + k) * 4;
        const int i0 = s_idx[e4 + 0], i1 = s_idx[e4 + 1];
        const int i2 = s_idx[e4 + 2], i3 = s_idx[e4 + 3];
        const float w0 = s_wt[e4 + 0], w1 = s_wt[e4 + 1];
        const float w2 = s_wt[e4 + 2], w3 = s_wt[e4 + 3];
        const float* xc = xb + ((size_t)(c0 + cl) << 14);
        union { _Float16 h[8]; uint4 u; } pk;
#pragma unroll
        for (int jj = 0; jj < 8; ++jj) {
            const float* xp = xc + ((size_t)jj << 14);
            pk.h[jj] = (_Float16)(w0 * xp[i0] + w1 * xp[i1] +
                                  w2 * xp[i2] + w3 * xp[i3]);
        }
        *reinterpret_cast<uint4*>(&s_a[buf][pp * AROW + cl]) = pk.u;
    };

    const int wv = tid >> 5, lane = tid & 31;
    const int nt = wv & 3, mt0 = wv >> 2;       // wave owns tiles mt0, mt0+2
    v8f acc0 = {}, acc1 = {};

    produce(0, 0);
    __syncthreads();

    for (int q = 0; q < NCHUNK; ++q) {
        const int buf = q & 1;
        // B fragment: one coalesced 32B load per lane, reused for 2 WMMAs.
        const uint4* bp = reinterpret_cast<const uint4*>(
            bfrag + (((size_t)(q * 4 + nt) * 32 + lane) << 4));
        union { uint4 u[2]; v16h v; } bf;
        bf.u[0] = bp[0];
        bf.u[1] = bp[1];
        // A fragments (16-bit 16x32 layout): lane<16 -> K {0..15};
        // lane>=16 -> K {16..31}. 80B row stride keeps b128 aligned.
        const int lo = (lane & 16) ? 16 : 0;
        const _Float16* ar0 = &s_a[buf][(mt0 * 16 + (lane & 15)) * AROW + lo];
        const _Float16* ar1 = ar0 + 32 * AROW;  // tile mt0+2
        union { uint4 u[2]; v16h v; } a0, a1;
        a0.u[0] = *reinterpret_cast<const uint4*>(ar0);
        a0.u[1] = *reinterpret_cast<const uint4*>(ar0 + 8);
        a1.u[0] = *reinterpret_cast<const uint4*>(ar1);
        a1.u[1] = *reinterpret_cast<const uint4*>(ar1 + 8);

        acc0 = __builtin_amdgcn_wmma_f32_16x16x32_f16(
            false, a0.v, false, bf.v, (short)0, acc0, false, false);
        acc1 = __builtin_amdgcn_wmma_f32_16x16x32_f16(
            false, a1.v, false, bf.v, (short)0, acc1, false, false);

        if (q + 1 < NCHUNK) produce(q + 1, buf ^ 1);
        __syncthreads();
    }

    // Store D: lane holds column o = nt*16 + lane%16; VGPR r -> row r(+8).
    const int o     = nt * 16 + (lane & 15);
    const int rbase = (lane & 16) ? 8 : 0;
    float* ob = out + (((size_t)(b * OCH + o)) << 14) + (mbase & (HWSZ - 1)) + rbase;
    float* op0 = ob + mt0 * 16;
    float* op1 = ob + (mt0 + 2) * 16;
    *reinterpret_cast<float4*>(op0)     = make_float4(acc0[0], acc0[1], acc0[2], acc0[3]);
    *reinterpret_cast<float4*>(op0 + 4) = make_float4(acc0[4], acc0[5], acc0[6], acc0[7]);
    *reinterpret_cast<float4*>(op1)     = make_float4(acc1[0], acc1[1], acc1[2], acc1[3]);
    *reinterpret_cast<float4*>(op1 + 4) = make_float4(acc1[4], acc1[5], acc1[6], acc1[7]);
}

// ---------------------------------------------------------------------------
extern "C" void kernel_launch(void* const* d_in, const int* in_sizes, int n_in,
                              void* d_out, int out_size, void* d_ws, size_t ws_size,
                              hipStream_t stream) {
    const float* x   = (const float*)d_in[0];   // (4,64,128,128)
    const float* wgt = (const float*)d_in[1];   // (64,64,3,3)
    const float* pgw = (const float*)d_in[2];   // (27,64,3,3)
    const float* pgb = (const float*)d_in[3];   // (27,)
    float* out = (float*)d_out;                 // (4,64,128,128)

    float*    offm  = (float*)d_ws;                                      // 65536*27 f32
    _Float16* bfrag = (_Float16*)((char*)d_ws + (size_t)65536 * 27 * 4); // 72 KB

    pg_conv_kernel <<<256,  256, 0, stream>>>(x, pgw, pgb, offm);
    pack_b_kernel  <<<144,  256, 0, stream>>>(wgt, bfrag);
    dcn_wmma_kernel<<<1024, 256, 0, stream>>>(x, offm, bfrag, out);
    (void)in_sizes; (void)n_in; (void)out_size; (void)ws_size;
}
